// Spectre_41729902248173
// MI455X (gfx1250) — matrix operand order repaired and checked by reading, not compiled, not run on hardware
//
#include <hip/hip_runtime.h>
#include <hip/hip_bf16.h>

typedef _Float16 f16;
typedef __attribute__((ext_vector_type(8)))  _Float16 v8h;
typedef __attribute__((ext_vector_type(16))) _Float16 v16h;
typedef __attribute__((ext_vector_type(8)))  float    v8f;

#define B_     4
#define N_     4096
#define D_     1024
#define H_     16
#define DH_    64
#define FREQ_  2049
#define ISQ2_  0.70710678118654752f

#ifndef __has_builtin
#define __has_builtin(x) 0
#endif
#if __has_builtin(__builtin_amdgcn_global_load_async_to_lds_b128) && \
    __has_builtin(__builtin_amdgcn_s_wait_asynccnt)
#define USE_ASYNC_LDS 1
typedef int v4i_ __attribute__((__vector_size__(16)));
typedef __attribute__((address_space(1))) v4i_* gptr_t;   // global int4*
typedef __attribute__((address_space(3))) v4i_* lptr_t;   // LDS int4*
#else
#define USE_ASYNC_LDS 0
#endif

// ---------------------------------------------------------------- conversions
__global__ __launch_bounds__(256) void cvt_f32_f16(const float* __restrict__ s,
                                                   f16* __restrict__ d, int n) {
    int i = blockIdx.x * 256 + threadIdx.x;
    int stride = gridDim.x * 256;
    for (; i < n; i += stride) d[i] = (f16)s[i];
}

// ------------------------------------------------- xbar[b,d] = mean_n x[b,n,d]
__global__ __launch_bounds__(256) void xbar_kernel(const float* __restrict__ x,
                                                   float* __restrict__ xbar) {
    int idx = blockIdx.x * 256 + threadIdx.x;      // 0..4095 = b*1024+d
    int b = idx >> 10, d = idx & 1023;
    const float* p = x + (size_t)b * N_ * D_ + d;
    float acc = 0.f;
    for (int n = 0; n < N_; ++n) acc += p[(size_t)n * D_];
    xbar[idx] = acc * (1.0f / N_);
}

// ------------------- bar[b,h,:] = LN( (xbar[b] @ Wq.T)[h*64 : h*64+64] )
__global__ __launch_bounds__(64) void bar_kernel(const float* __restrict__ xbar,
                                                 const float* __restrict__ Wq,
                                                 const float* __restrict__ ln_g,
                                                 const float* __restrict__ ln_b,
                                                 float* __restrict__ bar) {
    const int bh = blockIdx.x;            // b*16 + h
    const int b  = bh >> 4;
    const int j  = threadIdx.x;           // dh 0..63
    const int col = (bh & 15) * DH_ + j;
    const float* xr = xbar + b * D_;
    const float* wr = Wq + (size_t)col * D_;
    float acc = 0.f;
    for (int k = 0; k < D_; ++k) acc += xr[k] * wr[k];
    __shared__ float sh[64];
    sh[j] = acc; __syncthreads();
    float mu = 0.f;
    for (int t = 0; t < 64; ++t) mu += sh[t];
    mu *= (1.0f / 64.0f);
    __syncthreads();
    float dd = acc - mu;
    sh[j] = dd * dd; __syncthreads();
    float var = 0.f;
    for (int t = 0; t < 64; ++t) var += sh[t];
    var *= (1.0f / 64.0f);
    bar[bh * DH_ + j] = dd * rsqrtf(var + 1e-6f) * ln_g[j] + ln_b[j];
}

// ---------------------------------- hidden = silu(bar @ W{g,w}up.T)  (64x256)
__global__ __launch_bounds__(256) void hidden_kernel(const float* __restrict__ bar,
                                                     const float* __restrict__ Wgup,
                                                     const float* __restrict__ Wwup,
                                                     float* __restrict__ hg,
                                                     float* __restrict__ hw) {
    int idx = blockIdx.x * 256 + threadIdx.x;   // < 32768
    int which = idx >> 14;
    int li = idx & 16383;
    int r = li >> 8, c = li & 255;
    const float* W  = which ? Wwup : Wgup;
    const float* br = bar + r * DH_;
    const float* wr = W + c * DH_;
    float acc = 0.f;
    for (int k = 0; k < DH_; ++k) acc += br[k] * wr[k];
    float sv = acc / (1.0f + __expf(-acc));
    (which ? hw : hg)[li] = sv;
}

// ------------------- gate[r, 2k+ri] = hg@Wgdn.T ; s[r,n] = hw@Wwdn.T
#define NGR (64 * 2 * FREQ_)
#define NS  (64 * N_)
__global__ __launch_bounds__(256) void gates_kernel(const float* __restrict__ hg,
                                                    const float* __restrict__ hw,
                                                    const float* __restrict__ Wgdn,
                                                    const float* __restrict__ Wwdn,
                                                    float* __restrict__ gate,
                                                    float* __restrict__ s) {
    int idx = blockIdx.x * 256 + threadIdx.x;
    if (idx < NGR) {
        int r = idx / (2 * FREQ_), o = idx % (2 * FREQ_);
        const float* h = hg + r * 256;
        const float* w = Wgdn + (size_t)o * 256;
        float acc = 0.f;
        for (int k = 0; k < 256; ++k) acc += h[k] * w[k];
        gate[idx] = acc;
    } else if (idx < NGR + NS) {
        int li = idx - NGR;
        int r = li >> 12, o = li & 4095;
        const float* h = hw + r * 256;
        const float* w = Wwdn + (size_t)o * 256;
        float acc = 0.f;
        for (int k = 0; k < 256; ++k) acc += h[k] * w[k];
        s[li] = acc;
    }
}

// ------------------------------------------------ WMMA GEMM  C = A @ W^T
// A: MxK f16 row-major, W: NcxK f16 row-major, C: MxNc f32 row-major
// 256 threads = 8 waves; 128x128 C tile; KT=32; wave = 32x64 (2x4 wmma tiles)
// Double-buffered LDS; async global->LDS prefetch of next K-tile overlaps WMMA.
__global__ __launch_bounds__(256)
void gemm_nt_wmma(const f16* __restrict__ A, const f16* __restrict__ W,
                  float* __restrict__ C, int M, int Nc, int K) {
    __shared__ __align__(16) f16 lA[2][128][40];   // 32 K + 8 pad, 80B row
    __shared__ __align__(16) f16 lB[2][128][40];
    const int tid  = threadIdx.x;
    const int lane = tid & 31;
    const int wave = tid >> 5;
    const int bm = blockIdx.x * 128;
    const int bn = blockIdx.y * 128;
    const int wr = (wave >> 1) * 32;   // 0/32/64/96
    const int wc = (wave & 1) * 64;    // 0/64

    const v8f vzero = {0.f, 0.f, 0.f, 0.f, 0.f, 0.f, 0.f, 0.f};
    v8f acc[2][4];
#pragma unroll
    for (int i = 0; i < 2; ++i)
#pragma unroll
        for (int j = 0; j < 4; ++j) acc[i][j] = vzero;

    const int lrow = tid >> 1;          // 0..127
    const int lseg = (tid & 1) * 16;    // 0 or 16 (f16 elements)

    const int am = lane & 15;
    const int ak = (lane >> 4) * 8;    // A: lanes 0-15 -> K0-7/16-23, 16-31 -> K8-15/24-31
    const int bk = (lane >> 4) * 8;    // B: lanes 0-15 -> K0-7/16-23, 16-31 -> K8-15/24-31

    auto compute = [&](int buf) {
        v16h af[2], bf[4];
#pragma unroll
        for (int i = 0; i < 2; ++i) {
            v8h lo = *(const v8h*)&lA[buf][wr + i * 16 + am][ak];
            v8h hi = *(const v8h*)&lA[buf][wr + i * 16 + am][ak + 16];
            af[i] = __builtin_shufflevector(lo, hi, 0, 1, 2, 3, 4, 5, 6, 7,
                                            8, 9, 10, 11, 12, 13, 14, 15);
        }
#pragma unroll
        for (int j = 0; j < 4; ++j) {
            v8h lo = *(const v8h*)&lB[buf][wc + j * 16 + am][bk];
            v8h hi = *(const v8h*)&lB[buf][wc + j * 16 + am][bk + 16];
            bf[j] = __builtin_shufflevector(lo, hi, 0, 1, 2, 3, 4, 5, 6, 7,
                                            8, 9, 10, 11, 12, 13, 14, 15);
        }
#pragma unroll
        for (int i = 0; i < 2; ++i)
#pragma unroll
            for (int j = 0; j < 4; ++j)
                acc[i][j] = __builtin_amdgcn_wmma_f32_16x16x32_f16(
                    false, af[i], false, bf[j], (short)0, acc[i][j], false, false);
    };

#if USE_ASYNC_LDS
    auto prefetch = [&](int buf, int k0) {
        const f16* ga = A + (size_t)(bm + lrow) * K + k0 + lseg;
        const f16* gb = W + (size_t)(bn + lrow) * K + k0 + lseg;
        __builtin_amdgcn_global_load_async_to_lds_b128(
            (gptr_t)ga,       (lptr_t)&lA[buf][lrow][lseg],     0, 0);
        __builtin_amdgcn_global_load_async_to_lds_b128(
            (gptr_t)(ga + 8), (lptr_t)&lA[buf][lrow][lseg + 8], 0, 0);
        __builtin_amdgcn_global_load_async_to_lds_b128(
            (gptr_t)gb,       (lptr_t)&lB[buf][lrow][lseg],     0, 0);
        __builtin_amdgcn_global_load_async_to_lds_b128(
            (gptr_t)(gb + 8), (lptr_t)&lB[buf][lrow][lseg + 8], 0, 0);
    };
    const int niter = K >> 5;
    prefetch(0, 0);
    for (int i = 0; i < niter; ++i) {
        const int cur = i & 1;
        if (i + 1 < niter) {
            prefetch(cur ^ 1, (i + 1) << 5);     // DMA next tile during compute
            __builtin_amdgcn_s_wait_asynccnt(4); // in-order: current tile done
        } else {
            __builtin_amdgcn_s_wait_asynccnt(0);
        }
        __syncthreads();                          // all waves' tile in LDS
        compute(cur);
        __syncthreads();                          // reads done before overwrite
    }
#else
    for (int k0 = 0; k0 < K; k0 += 32) {
        const uint4* ga = (const uint4*)(A + (size_t)(bm + lrow) * K + k0 + lseg);
        const uint4* gb = (const uint4*)(W + (size_t)(bn + lrow) * K + k0 + lseg);
        uint4 a0 = ga[0], a1 = ga[1];
        uint4 b0 = gb[0], b1 = gb[1];
        __syncthreads();
        *(uint4*)&lA[0][lrow][lseg]     = a0;
        *(uint4*)&lA[0][lrow][lseg + 8] = a1;
        *(uint4*)&lB[0][lrow][lseg]     = b0;
        *(uint4*)&lB[0][lrow][lseg + 8] = b1;
        __syncthreads();
        compute(0);
    }
#endif

    // C layout: VGPR r -> rows r (lanes 0-15) / r+8 (lanes 16-31), col = lane&15
    const int cn = lane & 15;
    const int rh = (lane >> 4) * 8;
#pragma unroll
    for (int i = 0; i < 2; ++i)
#pragma unroll
        for (int j = 0; j < 4; ++j) {
            int row0 = bm + wr + i * 16 + rh;
            int col  = bn + wc + j * 16 + cn;
#pragma unroll
            for (int r = 0; r < 8; ++r)
                C[(size_t)(row0 + r) * Nc + col] = acc[i][j][r];
        }
}

// ---------------------------------------------- FFT / gate / iFFT / butterflies
__device__ __forceinline__ void fft4096(float2* buf, int tid, float sign) {
    for (int i = tid; i < 4096; i += 256) {       // 12-bit bit-reverse permute
        int j = (int)(__brev((unsigned)i) >> 20);
        if (i < j) { float2 t = buf[i]; buf[i] = buf[j]; buf[j] = t; }
    }
    __syncthreads();
    int shift = 0;
    for (int len = 2; len <= 4096; len <<= 1, ++shift) {
        const int half = len >> 1;
        const float ang0 = sign * 6.28318530717958647692f / (float)len;
        for (int t = tid; t < 2048; t += 256) {
            const int j  = t & (half - 1);
            const int i0 = ((t >> shift) << (shift + 1)) + j;
            const int i1 = i0 + half;
            float sn, cs;
            __sincosf(ang0 * (float)j, &sn, &cs);
            float2 u = buf[i0];
            float2 w = buf[i1];
            float2 tw = make_float2(w.x * cs - w.y * sn, w.x * sn + w.y * cs);
            buf[i0] = make_float2(u.x + tw.x, u.y + tw.y);
            buf[i1] = make_float2(u.x - tw.x, u.y - tw.y);
        }
        __syncthreads();
    }
}

// one block per (b,h,dh) column; blockIdx.x = ((b*16+h)*64+dh); col = low 10 bits
__global__ __launch_bounds__(256)
void fft_gate_kernel(const float* __restrict__ v, const float* __restrict__ gate,
                     const float* __restrict__ s, f16* __restrict__ vt) {
    __shared__ float2 buf[4096];    // 32 KB
    const int tid = threadIdx.x;
    const int col = blockIdx.x & 1023;   // h*64+dh
    const int b   = blockIdx.x >> 10;
    const int bh  = blockIdx.x >> 6;     // b*16+h
    const float* vcol = v + (size_t)b * N_ * D_ + col;

    for (int n = tid; n < 4096; n += 256) {
        buf[n].x = vcol[(size_t)n * D_];
        buf[n].y = 0.f;
    }
    __syncthreads();
    fft4096(buf, tid, -1.0f);                     // forward (unnormalized)

    const float* g = gate + (size_t)bh * (2 * FREQ_);
    for (int k = tid; k <= 2048; k += 256) {      // gate + Hermitian extension
        float2 X = buf[k];
        float gre = g[2 * k], gim = g[2 * k + 1];
        float2 Y = make_float2(X.x * gre - X.y * gim, X.x * gim + X.y * gre);
        buf[k] = Y;
        if (k >= 1 && k <= 2047) buf[4096 - k] = make_float2(Y.x, -Y.y);
    }
    __syncthreads();
    fft4096(buf, tid, 1.0f);                      // inverse (unnormalized)

    // vt in buf[n].x (scaled 1/N = ortho^2); w -> buf[.].y; then refine + add
    const float* sc = s + (size_t)bh * N_;
    const float scale = 1.0f / 4096.0f;
    for (int i = tid; i < 2048; i += 256) {
        float e = buf[2 * i].x * scale;
        float o = buf[2 * i + 1].x * scale;
        buf[2 * i].x = e;
        buf[2 * i + 1].x = o;
        buf[i].y        = (e + o) * ISQ2_ * sc[i];
        buf[i + 2048].y = (e - o) * ISQ2_ * sc[i + 2048];
    }
    __syncthreads();
    f16* outcol = vt + (size_t)b * N_ * D_ + col;
    for (int i = tid; i < 2048; i += 256) {
        float lo = buf[i].y, hi = buf[i + 2048].y;
        float r0 = (lo + hi) * ISQ2_;
        float r1 = (lo - hi) * ISQ2_;
        outcol[(size_t)(2 * i) * D_]     = (f16)(buf[2 * i].x + r0);
        outcol[(size_t)(2 * i + 1) * D_] = (f16)(buf[2 * i + 1].x + r1);
    }
}

// ---------------------------------------------------------------------- launch
extern "C" void kernel_launch(void* const* d_in, const int* in_sizes, int n_in,
                              void* d_out, int out_size, void* d_ws, size_t ws_size,
                              hipStream_t stream) {
    (void)in_sizes; (void)n_in; (void)out_size; (void)ws_size;
    const float* x    = (const float*)d_in[0];
    const float* Wq   = (const float*)d_in[1];
    const float* Wv   = (const float*)d_in[2];
    const float* Wo   = (const float*)d_in[3];
    const float* ln_g = (const float*)d_in[4];
    const float* ln_b = (const float*)d_in[5];
    const float* Wgup = (const float*)d_in[6];
    const float* Wgdn = (const float*)d_in[7];
    const float* Wwup = (const float*)d_in[8];
    const float* Wwdn = (const float*)d_in[9];
    float* out = (float*)d_out;

    char* ws = (char*)d_ws;
    size_t off = 0;
    auto alloc = [&](size_t bytes) -> void* {
        void* p = ws + off;
        off = (off + bytes + 255) & ~(size_t)255;
        return p;
    };
    float* xbar = (float*)alloc((size_t)B_ * D_ * 4);
    float* bar  = (float*)alloc((size_t)64 * 64 * 4);
    float* hg   = (float*)alloc((size_t)64 * 256 * 4);
    float* hw   = (float*)alloc((size_t)64 * 256 * 4);
    float* gate = (float*)alloc((size_t)64 * 2 * FREQ_ * 4);
    float* sbuf = (float*)alloc((size_t)64 * N_ * 4);
    f16*   x16  = (f16*)alloc((size_t)B_ * N_ * D_ * 2);
    f16*   wv16 = (f16*)alloc((size_t)D_ * D_ * 2);
    f16*   wo16 = (f16*)alloc((size_t)D_ * D_ * 2);
    float* v    = (float*)alloc((size_t)B_ * N_ * D_ * 4);
    f16*   vt16 = (f16*)alloc((size_t)B_ * N_ * D_ * 2);

    const int NTOT = B_ * N_ * D_;                  // 16,777,216
    cvt_f32_f16<<<4096, 256, 0, stream>>>(x, x16, NTOT);
    cvt_f32_f16<<<1024, 256, 0, stream>>>(Wv, wv16, D_ * D_);
    cvt_f32_f16<<<1024, 256, 0, stream>>>(Wo, wo16, D_ * D_);
    xbar_kernel<<<16, 256, 0, stream>>>(x, xbar);
    bar_kernel<<<64, 64, 0, stream>>>(xbar, Wq, ln_g, ln_b, bar);
    hidden_kernel<<<128, 256, 0, stream>>>(bar, Wgup, Wwup, hg, hw);
    gates_kernel<<<(NGR + NS + 255) / 256, 256, 0, stream>>>(hg, hw, Wgdn, Wwdn,
                                                             gate, sbuf);
    dim3 gg(B_ * N_ / 128, D_ / 128);               // (128, 8)
    gemm_nt_wmma<<<gg, 256, 0, stream>>>(x16, wv16, v, B_ * N_, D_, D_);
    fft_gate_kernel<<<B_ * H_ * DH_, 256, 0, stream>>>(v, gate, sbuf, vt16);
    gemm_nt_wmma<<<gg, 256, 0, stream>>>(vt16, wo16, out, B_ * N_, D_, D_);
}
